// SpKBGATModified_84696755077667
// MI455X (gfx1250) — matrix operand order repaired and checked by reading, not compiled, not run on hardware
//
#include <hip/hip_runtime.h>
#include <hip/hip_fp16.h>
#include <math.h>

typedef __attribute__((ext_vector_type(16))) _Float16 v16h;
typedef __attribute__((ext_vector_type(8)))  _Float16 v8h;
typedef __attribute__((ext_vector_type(8)))  float    v8f;

#define ALPHA_LRELU 0.2f

static __device__ __forceinline__ float wave_sum32(float v) {
#pragma unroll
  for (int off = 16; off > 0; off >>= 1) v += __shfl_xor(v, off, 32);
  return v;
}

// ---- row-wise L2 normalize, write f16 zero-padded to ldp columns ----
__global__ void k_l2norm_f16(const float* __restrict__ src, _Float16* __restrict__ dst,
                             int R, int C, int ldp) {
  int row  = blockIdx.x * (blockDim.x >> 5) + (threadIdx.x >> 5);
  int lane = threadIdx.x & 31;
  if (row >= R) return;
  const float* s = src + (size_t)row * C;
  float ss = 0.f;
  for (int c = lane; c < C; c += 32) { float v = s[c]; ss += v * v; }
  ss = wave_sum32(ss);
  float inv = 1.f / fmaxf(sqrtf(ss), 1e-12f);
  _Float16* d = dst + (size_t)row * ldp;
  for (int c = lane; c < ldp; c += 32)
    d[c] = (c < C) ? (_Float16)(s[c] * inv) : (_Float16)0.f;
}

// ---- weight -> f16 Bt[n][k], zero-padded to Rpad x ldp ----
// transpose==0: Bt[r][c] = src[r*src_ld + coff + c]   (rows of 'a', x @ a_slice.T)
// transpose==1: Bt[r][c] = src[c*src_ld + r]          (W is KxN, need W.T rows)
__global__ void k_conv_w(const float* __restrict__ src, int src_ld, int coff, int transpose,
                         _Float16* __restrict__ dst, int R, int C, int Rpad, int ldp) {
  long idx = (long)blockIdx.x * blockDim.x + threadIdx.x;
  long total = (long)Rpad * ldp;
  if (idx >= total) return;
  int r = (int)(idx / ldp);
  int c = (int)(idx - (long)r * ldp);
  float v = 0.f;
  if (r < R && c < C)
    v = transpose ? src[(size_t)c * src_ld + r] : src[(size_t)r * src_ld + coff + c];
  dst[idx] = (_Float16)v;
}

// ---- f32 (R x C, contiguous) -> f16 (R x ldp, zero padded) ----
__global__ void k_conv_a(const float* __restrict__ src, _Float16* __restrict__ dst,
                         int R, int C, int ldp) {
  long idx = (long)blockIdx.x * blockDim.x + threadIdx.x;
  long total = (long)R * ldp;
  if (idx >= total) return;
  int r = (int)(idx / ldp);
  int c = (int)(idx - (long)r * ldp);
  dst[idx] = (c < C) ? (_Float16)src[(size_t)r * C + c] : (_Float16)0.f;
}

// ---- WMMA GEMM, register-blocked over the full N-tile row ----
// C(MxN,f32) = A(MxKpad,f16,lda) * Bt(n-major: Bt[n][k], ldb), N <= NT*16.
// Block = 8 waves = 8 consecutive M-tiles. Each wave keeps NT v8f accumulators.
// Per 32-wide K step: B chunk (NT*16 rows x 32 halfs) is staged to LDS once per
// block (rows padded to 40 halfs), then each wave issues NT v_wmma on one A frag.
// Fragment layouts per CDNA5 ISA 7.12.2:
//  A (16x32 f16): lane l -> row l%16, g=l/16; two 16B runs at k0+8g, k0+16+8g.
//  B (32x16 f16): lane l -> col l%16, g=l/16; two 16B runs at k0+8g, k0+16+8g.
template <int NT>
__global__ void k_gemm_wmma(const _Float16* __restrict__ A, int lda,
                            const _Float16* __restrict__ B, int ldb,
                            float* __restrict__ C, int ldc,
                            int M, int N, int Kpad, int mt) {
  const int BROW = 40;  // padded LDS row stride (halfs), 16B aligned
  __shared__ _Float16 Bs[NT * 16 * BROW];
  int lane = threadIdx.x & 31;
  int tm = blockIdx.x * 8 + (threadIdx.x >> 5);
  int g = lane >> 4;
  int p = lane & 15;
  int arow = tm * 16 + p;
  bool avalid = (tm < mt) && (arow < M);
  const _Float16* Ap = A + (size_t)(avalid ? arow : 0) * lda + g * 8;
  v8f acc[NT] = {};
  v16h azero = {};
  for (int k0 = 0; k0 < Kpad; k0 += 32) {
    // cooperative stage of B K-chunk
    for (int idx = threadIdx.x; idx < NT * 16 * 4; idx += 256) {
      int row = idx >> 2, seg = idx & 3;
      *(v8h*)(Bs + row * BROW + seg * 8) =
          *(const v8h*)(B + (size_t)row * ldb + k0 + seg * 8);
    }
    __syncthreads();
    union { v16h v; v8h h[2]; } a;
    if (avalid) {
      a.h[0] = *(const v8h*)(Ap + k0);
      a.h[1] = *(const v8h*)(Ap + k0 + 16);
      if (k0 + 32 < Kpad) __builtin_prefetch(Ap + k0 + 32, 0, 3);
    } else {
      a.v = azero;
    }
#pragma unroll
    for (int t = 0; t < NT; ++t) {
      union { v16h v; v8h h[2]; } b;
      const _Float16* bp = Bs + (t * 16 + p) * BROW + g * 8;
      b.h[0] = *(const v8h*)(bp);
      b.h[1] = *(const v8h*)(bp + 16);
      acc[t] = __builtin_amdgcn_wmma_f32_16x16x32_f16(false, a.v, false, b.v,
                                                      (short)0, acc[t], false, false);
    }
    __syncthreads();
  }
  if (tm >= mt) return;
  int rbase = tm * 16 + g * 8;
#pragma unroll
  for (int t = 0; t < NT; ++t) {
    int col = t * 16 + p;
    if (col < N) {
#pragma unroll
      for (int v = 0; v < 8; ++v) {
        int row = rbase + v;
        if (row < M) C[(size_t)row * ldc + col] = acc[t][v];
      }
    }
  }
}

// ---- per-row dot with a2: out[r] = sum_c x[r][c]*a2[c] ----
__global__ void k_rowdot(const float* __restrict__ x, const float* __restrict__ a2,
                         float* __restrict__ out, int R, int C) {
  int row  = blockIdx.x * (blockDim.x >> 5) + (threadIdx.x >> 5);
  int lane = threadIdx.x & 31;
  if (row >= R) return;
  const float* px = x + (size_t)row * C;
  float s = 0.f;
  for (int c = lane; c < C; c += 32) s += px[c] * a2[c];
  s = wave_sum32(s);
  if (lane == 0) out[row] = s;
}

__global__ void k_mask_set(const int* __restrict__ batch, float* __restrict__ mask, int n) {
  int i = blockIdx.x * blockDim.x + threadIdx.x;
  if (i < n) mask[batch[(size_t)i * 3 + 2]] = 1.0f;
}

// ---- edge pass: one wave per edge (E normal edges then EN n-hop edges) ----
__global__ void k_edge(const int* __restrict__ esrc, const int* __restrict__ edst,
                       const int* __restrict__ etype, const int* __restrict__ nhop,
                       const float* __restrict__ ssrc, const float* __restrict__ sdst,
                       const float* __restrict__ srel,
                       const float* __restrict__ xd, const float* __restrict__ rp,
                       float* __restrict__ rowsum, float* __restrict__ hacc,
                       int E, int EN, int din) {
  int wave = (int)((blockIdx.x * blockDim.x + threadIdx.x) >> 5);
  int lane = (int)(threadIdx.x & 31);
  if (wave >= E + EN) return;
  int s, d;
  const float* r0;
  const float* r1 = nullptr;
  float rs;
  if (wave < E) {
    s = esrc[wave]; d = edst[wave];
    int t = etype[wave];
    rs = srel[t];
    r0 = rp + (size_t)t * din;
  } else {
    const int* row = nhop + (size_t)(wave - E) * 4;
    d = row[0]; int i = row[1]; int j = row[2]; s = row[3];
    rs = srel[i] + srel[j];
    r0 = rp + (size_t)i * din;
    r1 = rp + (size_t)j * din;
  }
  float logit = ssrc[s] + sdst[d] + rs;
  float lr = logit > 0.f ? logit : ALPHA_LRELU * logit;
  float w = expf(-lr);
  if (lane == 0) atomicAdd(rowsum + s, w);
  const float* xr = xd + (size_t)d * din;
  float* acc = hacc + (size_t)s * din;
  for (int c = lane; c < din; c += 32) {
    float v = xr[c] + r0[c] + (r1 ? r1[c] : 0.f);
    atomicAdd(acc + c, w * v);
  }
}

// ---- finalize: h' = elu((hacc + rowsum*xs)/denom); optional f32 and f16 outputs ----
__global__ void k_finalize(const float* __restrict__ hacc, const float* __restrict__ xs,
                           const float* __restrict__ rowsum, int Nn, int din,
                           float* __restrict__ dst32, int ld32,
                           _Float16* __restrict__ dst16, int ld16, int coff) {
  long idx = (long)blockIdx.x * blockDim.x + threadIdx.x;
  long total = (long)Nn * din;
  if (idx >= total) return;
  int n = (int)(idx / din);
  int c = (int)(idx - (long)n * din);
  float rs = rowsum[n];
  float denom = (rs == 0.f) ? 1e-12f : rs;
  float v = (hacc[idx] + rs * xs[idx]) / denom;
  v = v > 0.f ? v : (expf(v) - 1.f);           // elu
  if (dst32) dst32[(size_t)n * ld32 + c] = v;
  if (dst16) dst16[(size_t)n * ld16 + coff + c] = (_Float16)v;
}

// ---- combine + l2norm: out = l2norm(base + m*add); optional f16 copy (padded) ----
__global__ void k_combine(const float* __restrict__ base, const float* __restrict__ add,
                          const float* __restrict__ mask,
                          float* __restrict__ out32,
                          _Float16* __restrict__ out16, int ld16,
                          int R, int C) {
  int row  = blockIdx.x * (blockDim.x >> 5) + (threadIdx.x >> 5);
  int lane = threadIdx.x & 31;
  if (row >= R) return;
  float m = mask ? mask[row] : 1.0f;
  const float* b = base + (size_t)row * C;
  const float* a = add + (size_t)row * C;
  float tv[8];
  float ss = 0.f;
  int k = 0;
  for (int c = lane; c < C; c += 32) { float t = b[c] + m * a[c]; tv[k++] = t; ss += t * t; }
  ss = wave_sum32(ss);
  float inv = 1.f / fmaxf(sqrtf(ss), 1e-12f);
  k = 0;
  for (int c = lane; c < C; c += 32) {
    float o = tv[k++] * inv;
    out32[(size_t)row * C + c] = o;
    if (out16) out16[(size_t)row * ld16 + c] = (_Float16)o;
  }
  if (out16)
    for (int c = C + lane; c < ld16; c += 32) out16[(size_t)row * ld16 + c] = (_Float16)0.f;
}

static inline unsigned gblocks(long total, int tpb) {
  return (unsigned)((total + tpb - 1) / tpb);
}

extern "C" void kernel_launch(void* const* d_in, const int* in_sizes, int n_in,
                              void* d_out, int out_size, void* d_ws, size_t ws_size,
                              hipStream_t stream) {
  (void)n_in; (void)out_size; (void)ws_size;
  const float* entity   = (const float*)d_in[0];
  const float* relation = (const float*)d_in[1];
  const float* W_ent    = (const float*)d_in[2];
  const float* g0_a0    = (const float*)d_in[3];
  const float* g0_s0    = (const float*)d_in[4];
  const float* g0_a1    = (const float*)d_in[5];
  const float* g0_s1    = (const float*)d_in[6];
  const float* g0_Wm    = (const float*)d_in[7];
  const float* g0_oa    = (const float*)d_in[8];
  const float* g0_os    = (const float*)d_in[9];
  const float* g1_a0    = (const float*)d_in[10];
  const float* g1_s0    = (const float*)d_in[11];
  const float* g1_Wm    = (const float*)d_in[12];
  const float* g1_oa    = (const float*)d_in[13];
  const float* g1_os    = (const float*)d_in[14];
  const int* edge_list  = (const int*)d_in[15];
  const int* edge_type  = (const int*)d_in[16];
  const int* nhop       = (const int*)d_in[17];
  const int* batch      = (const int*)d_in[18];

  const int D0 = 100, D1 = 200, KP0 = 128, KP1 = 224;
  const int NN = in_sizes[0] / D0;   // 50000 nodes
  const int NR = in_sizes[1] / D0;   // 500 relations
  const int E  = in_sizes[15] / 2;   // 500000 edges
  const int EN = in_sizes[17] / 4;   // 100000 n-hop edges
  const int NB = in_sizes[18] / 3;   // 10000 batch rows
  const int* esrc = edge_list;
  const int* edst = edge_list + E;

  // ---- workspace arena ----
  char* wsb = (char*)d_ws;
  size_t off = 0;
  auto carve = [&](size_t bytes) -> void* {
    void* p = wsb + off;
    off += (bytes + 255) & ~(size_t)255;
    return p;
  };
  float* ent_up = (float*)carve((size_t)NN * D1 * 4);
  float* xs     = (float*)carve((size_t)NN * D1 * 4);
  float* xd     = (float*)carve((size_t)NN * D1 * 4);
  float* hacc   = (float*)carve((size_t)NN * D1 * 4);
  float* ent1   = (float*)carve((size_t)NN * D1 * 4);
  float* enew   = (float*)carve((size_t)NN * D1 * 4);
  float* rowsum = (float*)carve((size_t)NN * 4);
  float* ssrc   = (float*)carve((size_t)NN * 4);
  float* sdst   = (float*)carve((size_t)NN * 4);
  float* maskb  = (float*)carve((size_t)NN * 4);
  float* rel_up = (float*)carve((size_t)NR * D1 * 4);
  float* rp     = (float*)carve((size_t)NR * D1 * 4);
  float* rnew   = (float*)carve((size_t)NR * D1 * 4);
  float* rel1   = (float*)carve((size_t)NR * D1 * 4);
  float* srel   = (float*)carve((size_t)NR * 4);
  _Float16* AhA = (_Float16*)carve((size_t)NN * KP1 * 2);
  _Float16* AhB = (_Float16*)carve((size_t)NN * KP1 * 2);
  _Float16* RhA = (_Float16*)carve((size_t)NR * KP1 * 2);
  _Float16* RhB = (_Float16*)carve((size_t)NR * KP1 * 2);
  _Float16* Wt  = (_Float16*)carve((size_t)208 * KP1 * 2);

  const int TPB = 256, RPB = 8;  // 8 waves per block for wave-per-row kernels

  auto conv_w = [&](const float* src, int src_ld, int coff, int transpose,
                    int R, int C, int Kpad) {
    int Rpad = ((R + 15) / 16) * 16;
    long total = (long)Rpad * Kpad;
    k_conv_w<<<gblocks(total, TPB), TPB, 0, stream>>>(src, src_ld, coff, transpose,
                                                      Wt, R, C, Rpad, Kpad);
  };
  auto gemm = [&](const _Float16* A, int lda, float* C, int M, int N, int Kpad) {
    int mt = (M + 15) / 16, nt = (N + 15) / 16;
    unsigned blocks = (unsigned)((mt + 7) / 8);
    if (nt == 7)
      k_gemm_wmma<7><<<blocks, 256, 0, stream>>>(A, lda, Wt, Kpad, C, N, M, N, Kpad, mt);
    else
      k_gemm_wmma<13><<<blocks, 256, 0, stream>>>(A, lda, Wt, Kpad, C, N, M, N, Kpad, mt);
  };
  // one factored sparse-attention pass (always followed by elu per reference)
  auto attention = [&](const _Float16* Axh, const _Float16* Rxh, int lda,
                       const float* a, int a_ld, const float* a2,
                       int din, int Kpad,
                       float* fin32, _Float16* fin16, int fcoff) {
    conv_w(a, a_ld, 0, 0, din, din, Kpad);        gemm(Axh, lda, xs, NN, din, Kpad);
    conv_w(a, a_ld, din, 0, din, din, Kpad);      gemm(Axh, lda, xd, NN, din, Kpad);
    conv_w(a, a_ld, 2 * din, 0, din, din, Kpad);  gemm(Rxh, lda, rp, NR, din, Kpad);
    k_rowdot<<<gblocks(NN, RPB), TPB, 0, stream>>>(xs, a2, ssrc, NN, din);
    k_rowdot<<<gblocks(NN, RPB), TPB, 0, stream>>>(xd, a2, sdst, NN, din);
    k_rowdot<<<gblocks(NR, RPB), TPB, 0, stream>>>(rp, a2, srel, NR, din);
    hipMemsetAsync(rowsum, 0, (size_t)NN * 4, stream);
    hipMemsetAsync(hacc, 0, (size_t)NN * din * 4, stream);
    long ethreads = (long)(E + EN) * 32;
    k_edge<<<gblocks(ethreads, TPB), TPB, 0, stream>>>(esrc, edst, edge_type, nhop,
        ssrc, sdst, srel, xd, rp, rowsum, hacc, E, EN, din);
    long ftotal = (long)NN * din;
    k_finalize<<<gblocks(ftotal, TPB), TPB, 0, stream>>>(hacc, xs, rowsum, NN, din,
        fin32, din, fin16, KP1, fcoff);
  };

  // ---- phase 0: l2norm inputs, mask, ent_up / rel_up ----
  k_l2norm_f16<<<gblocks(NN, RPB), TPB, 0, stream>>>(entity, AhA, NN, D0, KP0);
  k_l2norm_f16<<<gblocks(NR, RPB), TPB, 0, stream>>>(relation, RhA, NR, D0, KP0);
  hipMemsetAsync(maskb, 0, (size_t)NN * 4, stream);
  k_mask_set<<<gblocks(NB, TPB), TPB, 0, stream>>>(batch, maskb, NB);
  hipMemsetAsync(AhB, 0, (size_t)NN * KP1 * 2, stream);  // pad cols for layer-1 K
  conv_w(W_ent, D1, 0, 1, D1, D0, KP0);
  gemm(AhA, KP0, ent_up, NN, D1, KP0);
  gemm(RhA, KP0, rel_up, NR, D1, KP0);

  // ---- phase 1: layer-0 heads -> x0 (f16, directly into AhB halves) ----
  attention(AhA, RhA, KP0, g0_a0, 3 * D0, g0_s0, D0, KP0, nullptr, AhB, 0);
  attention(AhA, RhA, KP0, g0_a1, 3 * D0, g0_s1, D0, KP0, nullptr, AhB, D0);

  // ---- phase 2: layer-0 out attention + residual updates ----
  conv_w(g0_Wm, D1, 0, 1, D1, D0, KP0);
  gemm(RhA, KP0, rnew, NR, D1, KP0);               // r_new0 = rel @ g0_W
  k_conv_a<<<gblocks((long)NR * KP1, TPB), TPB, 0, stream>>>(rnew, RhB, NR, D1, KP1);
  attention(AhB, RhB, KP1, g0_oa, 3 * D1, g0_os, D1, KP1, enew, nullptr, 0);
  k_combine<<<gblocks(NN, RPB), TPB, 0, stream>>>(ent_up, enew, maskb, ent1, AhA, KP1, NN, D1);
  k_combine<<<gblocks(NR, RPB), TPB, 0, stream>>>(rel_up, rnew, nullptr, rel1, RhA, KP1, NR, D1);

  // ---- phase 3: layer-1 head -> x1 (f16 into AhB; pad cols still zero) ----
  attention(AhA, RhA, KP1, g1_a0, 3 * D1, g1_s0, D1, KP1, nullptr, AhB, 0);

  // ---- phase 4: layer-1 out attention + final residual + outputs ----
  conv_w(g1_Wm, D1, 0, 1, D1, D1, KP1);
  gemm(RhA, KP1, rnew, NR, D1, KP1);               // r_new1 = rel1 @ g1_W
  k_conv_a<<<gblocks((long)NR * KP1, TPB), TPB, 0, stream>>>(rnew, RhB, NR, D1, KP1);
  attention(AhB, RhB, KP1, g1_oa, 3 * D1, g1_os, D1, KP1, enew, nullptr, 0);

  float* out_ent = (float*)d_out;
  float* out_rel = out_ent + (size_t)NN * D1;
  k_combine<<<gblocks(NN, RPB), TPB, 0, stream>>>(ent1, enew, maskb, out_ent,
                                                  (_Float16*)nullptr, 0, NN, D1);
  k_combine<<<gblocks(NR, RPB), TPB, 0, stream>>>(rel1, rnew, nullptr, out_rel,
                                                  (_Float16*)nullptr, 0, NR, D1);
}
